// ResidualBlock_4063039062847
// MI455X (gfx1250) — matrix operand order repaired and checked by reading, not compiled, not run on hardware
//
#include <hip/hip_runtime.h>

// ---------------------------------------------------------------------------
// ResidualBlock (sparse conv -> BN -> ReLU -> sparse conv -> BN -> +skip -> ReLU)
// MI455X / gfx1250: bf16 WMMA (v_wmma_f32_16x16x32_bf16), f32 accumulation.
// Weight slices staged with global_load_async_to_lds_b128 (ASYNCcnt),
// double-buffered per koff; transposed gather indices for coalescing.
// ---------------------------------------------------------------------------

#define NN 393216
#define CC 64
#define KK 27
#define EPSV 1e-5f

typedef __attribute__((ext_vector_type(16))) __bf16 v16bf;
typedef __attribute__((ext_vector_type(8)))  float  v8f;

union BFrag { uint4 u[2]; v16bf v; };

__device__ __forceinline__ unsigned short f32_to_bf16(float f) {
    unsigned int u = __float_as_uint(f);
    u += 0x7fffu + ((u >> 16) & 1u);          // round-to-nearest-even
    return (unsigned short)(u >> 16);
}
__device__ __forceinline__ float bf16_to_f32(unsigned short s) {
    return __uint_as_float(((unsigned int)s) << 16);
}

// --- zero the global BN-stat accumulators (512 floats) ----------------------
__global__ void zero_stats_kernel(float* stats) {
    stats[blockIdx.x * 256 + threadIdx.x] = 0.0f;
}

// --- feats f32 -> bf16 ------------------------------------------------------
__global__ __launch_bounds__(256) void cvt_bf16_kernel(const float* __restrict__ x,
                                                       unsigned short* __restrict__ y,
                                                       int n) {
    int i = blockIdx.x * 256 + threadIdx.x;
    if (i < n) y[i] = f32_to_bf16(x[i]);
}

// --- transpose nbr [N][27] -> nbrT [27][N] for coalesced per-koff reads -----
__global__ __launch_bounds__(256) void transpose_nbr_kernel(const int* __restrict__ nbr,
                                                            int* __restrict__ nbrT) {
    int row  = blockIdx.x * 256 + threadIdx.x;
    int koff = blockIdx.y;
    nbrT[koff * NN + row] = nbr[row * KK + koff];
}

// --- pack W [27][64][64] f32 into per-lane WMMA B-fragment layout (bf16) ----
// Bpack index = koff*4096 + (kc*4+nt)*512 + lane*16 + e
// lane -> n = nt*16 + (lane&15); k = kc*32 + (lane>>4)*16 + e   (e = 0..15)
__global__ __launch_bounds__(256) void pack_w_kernel(const float* __restrict__ W,
                                                     unsigned short* __restrict__ Bp) {
    int i = blockIdx.x * 256 + threadIdx.x;
    if (i >= KK * 2 * 4 * 32 * 16) return;
    int e    = i & 15;
    int lane = (i >> 4) & 31;
    int nt   = (i >> 9) & 3;
    int kc   = (i >> 11) & 1;
    int koff = i >> 12;
    int n = nt * 16 + (lane & 15);
    int k = kc * 32 + (lane >> 4) * 16 + e;
    Bp[i] = f32_to_bf16(W[(koff * CC + k) * CC + n]);
}

// --- gather + GEMM (WMMA) + scatter sparse conv, with per-channel stats -----
// One wave computes a 64x64 output tile (rows [waveid*64, +64), all 64 cols).
// The 8 KB weight slice for koff+1 is fetched with async-to-LDS loads while
// the current slice feeds the WMMAs; s_wait_asynccnt + barrier publishes it.
template <bool OUT_F32>
__global__ __launch_bounds__(256) void spconv_wmma_kernel(
    const unsigned short* __restrict__ xbf,   // [N,64] bf16 input features
    const unsigned short* __restrict__ Bp,    // packed weights (bf16 fragments)
    const int*            __restrict__ nbrT,  // [27,N] gather indices
    float*                __restrict__ out32, // [N,64] f32 (conv2)
    unsigned short*       __restrict__ outbf, // [N,64] bf16 (conv1)
    float* __restrict__ gsum, float* __restrict__ gsq)
{
    __shared__ unsigned short Bs[2][4096];    // 2 x 8 KB weight slices
    __shared__ float ssum[CC];
    __shared__ float ssq[CC];

    const int tid  = threadIdx.x;
    const int lane = tid & 31;
    const int wave = tid >> 5;
    const int m16  = lane & 15;
    const int h    = lane >> 4;
    const int rowbase = (blockIdx.x * 8 + wave) * 64;

    if (tid < CC) { ssum[tid] = 0.f; ssq[tid] = 0.f; }

    v8f acc[4][4];
#pragma unroll
    for (int mt = 0; mt < 4; ++mt)
#pragma unroll
        for (int nt = 0; nt < 4; ++nt)
            acc[mt][nt] = (v8f){0.f, 0.f, 0.f, 0.f, 0.f, 0.f, 0.f, 0.f};

    // prologue: async-stage koff = 0 slice into Bs[0] (32 B per thread)
    {
        unsigned int l0 = (unsigned int)(size_t)&Bs[0][tid * 16];   // LDS byte offset
        unsigned int g0 = (unsigned int)(tid * 32);                 // global byte offset
        asm volatile(
            "global_load_async_to_lds_b128 %0, %1, %2\n\t"
            "global_load_async_to_lds_b128 %3, %4, %2"
            :: "v"(l0), "v"(g0), "s"(Bp), "v"(l0 + 16u), "v"(g0 + 16u)
            : "memory");
        asm volatile("s_wait_asynccnt 0" ::: "memory");
    }
    __syncthreads();

    for (int koff = 0; koff < KK; ++koff) {
        const int  cur  = koff & 1;
        const bool have = (koff + 1 < KK);

        // issue async staging of the next slice (hidden behind compute)
        if (have) {
            unsigned int l0 = (unsigned int)(size_t)&Bs[cur ^ 1][tid * 16];
            unsigned int g0 = (unsigned int)((koff + 1) * 8192 + tid * 32);
            asm volatile(
                "global_load_async_to_lds_b128 %0, %1, %2\n\t"
                "global_load_async_to_lds_b128 %3, %4, %2"
                :: "v"(l0), "v"(g0), "s"(Bp), "v"(l0 + 16u), "v"(g0 + 16u)
                : "memory");
        }

        // coalesced per-koff index loads
        int idx[4];
#pragma unroll
        for (int mt = 0; mt < 4; ++mt)
            idx[mt] = nbrT[koff * NN + rowbase + mt * 16 + m16];

#pragma unroll
        for (int kc = 0; kc < 2; ++kc) {
            // B fragments from LDS: 2x ds_load_b128 per fragment
            BFrag b[4];
#pragma unroll
            for (int nt = 0; nt < 4; ++nt) {
                const uint4* p = (const uint4*)&Bs[cur][((kc * 4 + nt) * 32 + lane) * 16];
                b[nt].u[0] = p[0];
                b[nt].u[1] = p[1];
            }
#pragma unroll
            for (int mt = 0; mt < 4; ++mt) {
                // A fragment gather: row idx[mt], 16-bit A 16x32 lane layout:
                // elements kc*32 + 16h + [0,16) (contiguous, matches packed B)
                const unsigned short* rp = xbf + idx[mt] * CC + kc * 32 + h * 16;
                BFrag a;
                a.u[0] = *(const uint4*)(rp);
                a.u[1] = *(const uint4*)(rp + 8);
#pragma unroll
                for (int nt = 0; nt < 4; ++nt)
                    acc[mt][nt] = __builtin_amdgcn_wmma_f32_16x16x32_bf16(
                        false, a.v, false, b[nt].v,
                        (short)0, acc[mt][nt], false, false);
            }
        }

        // publish next slice: drain async writes, then barrier
        if (have) asm volatile("s_wait_asynccnt 0" ::: "memory");
        __syncthreads();
    }

    // --- scatter results + per-channel BN statistics -------------------------
#pragma unroll
    for (int nt = 0; nt < 4; ++nt) {
        const int c = nt * 16 + m16;           // C/D layout: col = lane%16
        float s = 0.f, q = 0.f;
#pragma unroll
        for (int mt = 0; mt < 4; ++mt) {
#pragma unroll
            for (int r = 0; r < 8; ++r) {      // C/D layout: row = 8*h + r
                float v = acc[mt][nt][r];
                s += v; q += v * v;
                int row = rowbase + mt * 16 + h * 8 + r;
                if (OUT_F32) out32[row * CC + c] = v;
                else         outbf[row * CC + c] = f32_to_bf16(v);
            }
        }
        atomicAdd(&ssum[c], s);                // ds_add_f32
        atomicAdd(&ssq[c],  q);
    }
    __syncthreads();
    if (tid < CC) {
        atomicAdd(&gsum[tid], ssum[tid]);      // global_atomic_add_f32
        atomicAdd(&gsq[tid],  ssq[tid]);
    }
}

// --- BN finalize: per-channel scale/shift -----------------------------------
__global__ void bn_finalize_kernel(const float* __restrict__ gsum,
                                   const float* __restrict__ gsq,
                                   const float* __restrict__ gamma,
                                   const float* __restrict__ beta,
                                   float* __restrict__ scale,
                                   float* __restrict__ shift) {
    int c = threadIdx.x;
    float inv_n = 1.0f / (float)NN;
    float mu  = gsum[c] * inv_n;
    float var = gsq[c] * inv_n - mu * mu;      // biased variance
    float sc  = gamma[c] * rsqrtf(var + EPSV);
    scale[c] = sc;
    shift[c] = beta[c] - mu * sc;
}

// --- BN + ReLU, bf16 in place (conv1 out -> conv2 in) -----------------------
__global__ __launch_bounds__(256) void bnrelu_bf16_kernel(unsigned short* __restrict__ h,
                                                          const float* __restrict__ scale,
                                                          const float* __restrict__ shift,
                                                          int n) {
    int i = blockIdx.x * 256 + threadIdx.x;
    if (i >= n) return;
    int c = i & (CC - 1);
    float v = bf16_to_f32(h[i]) * scale[c] + shift[c];
    h[i] = f32_to_bf16(fmaxf(v, 0.0f));
}

// --- final: BN2 + identity skip + ReLU, f32 in place on d_out ---------------
__global__ __launch_bounds__(256) void bn_skip_relu_kernel(float* __restrict__ out,
                                                           const float* __restrict__ feats,
                                                           const float* __restrict__ scale,
                                                           const float* __restrict__ shift,
                                                           int n) {
    int i = blockIdx.x * 256 + threadIdx.x;
    if (i >= n) return;
    int c = i & (CC - 1);
    float v = out[i] * scale[c] + shift[c] + feats[i];
    out[i] = fmaxf(v, 0.0f);
}

// ---------------------------------------------------------------------------
extern "C" void kernel_launch(void* const* d_in, const int* in_sizes, int n_in,
                              void* d_out, int out_size, void* d_ws, size_t ws_size,
                              hipStream_t stream) {
    const float* feats  = (const float*)d_in[0];
    const int*   nbr    = (const int*)  d_in[1];
    const float* W1     = (const float*)d_in[2];
    const float* W2     = (const float*)d_in[3];
    const float* gamma1 = (const float*)d_in[4];
    const float* beta1  = (const float*)d_in[5];
    const float* gamma2 = (const float*)d_in[6];
    const float* beta2  = (const float*)d_in[7];
    float* out = (float*)d_out;

    // workspace layout
    char* ws = (char*)d_ws;
    const size_t nfeat = (size_t)NN * CC;                   // 25,165,824 elems
    const int    packN = KK * 2 * 4 * 32 * 16;              // 110,592 elems
    unsigned short* featsbf = (unsigned short*)ws;                       // 50 MB
    unsigned short* hbf     = (unsigned short*)(ws + nfeat * 2);         // 50 MB
    unsigned short* Bp1     = (unsigned short*)(ws + nfeat * 4);         // 216 KB
    unsigned short* Bp2     = Bp1 + packN;                               // 216 KB
    int*            nbrT    = (int*)(Bp2 + packN);                       // 42 MB
    float* stats = (float*)(nbrT + (size_t)KK * NN);
    float* sum1 = stats;        float* sq1 = stats + 64;
    float* sum2 = stats + 128;  float* sq2 = stats + 192;
    float* scale1 = stats + 256; float* shift1 = stats + 320;
    float* scale2 = stats + 384; float* shift2 = stats + 448;

    const int elemBlocks = (int)(nfeat / 256);              // 98304
    const int packBlk    = (packN + 255) / 256;
    const int convBlocks = NN / 512;                        // 8 waves * 64 rows

    zero_stats_kernel<<<2, 256, 0, stream>>>(stats);        // 512 floats
    cvt_bf16_kernel<<<elemBlocks, 256, 0, stream>>>(feats, featsbf, (int)nfeat);
    transpose_nbr_kernel<<<dim3(NN / 256, KK), 256, 0, stream>>>(nbr, nbrT);
    pack_w_kernel<<<packBlk, 256, 0, stream>>>(W1, Bp1);
    pack_w_kernel<<<packBlk, 256, 0, stream>>>(W2, Bp2);

    // conv1: bf16 out + stats
    spconv_wmma_kernel<false><<<convBlocks, 256, 0, stream>>>(
        featsbf, Bp1, nbrT, nullptr, hbf, sum1, sq1);
    bn_finalize_kernel<<<1, 64, 0, stream>>>(sum1, sq1, gamma1, beta1, scale1, shift1);
    bnrelu_bf16_kernel<<<elemBlocks, 256, 0, stream>>>(hbf, scale1, shift1, (int)nfeat);

    // conv2: f32 out into d_out + stats
    spconv_wmma_kernel<true><<<convBlocks, 256, 0, stream>>>(
        hbf, Bp2, nbrT, out, nullptr, sum2, sq2);
    bn_finalize_kernel<<<1, 64, 0, stream>>>(sum2, sq2, gamma2, beta2, scale2, shift2);
    bn_skip_relu_kernel<<<elemBlocks, 256, 0, stream>>>(out, feats, scale2, shift2, (int)nfeat);
}
